// CelltypeDeconvolver_71700184039973
// MI455X (gfx1250) — compile-verified, matrix operand order checked by reading, not run on hardware
//
#include <hip/hip_runtime.h>
#include <hip/hip_bf16.h>

// ---------------------------------------------------------------------------
// CelltypeDeconvolver for MI455X (gfx1250, wave32, WMMA bf16 matrix pipe)
// All GEMMs: v_wmma_f32_16x16x32_bf16, branch-free b128 fragment loads,
// scalar (SGPR) tile bases via readfirstlane, bf16 activations kept
// zero-padded so no guards exist anywhere in the K loops. bf16 copy of x is
// L2-resident (100 MB < 192 MB) across the 8 re-reads by the n-group waves.
// global_prefetch_b8 pulls the next K block while wmma drains.
// ---------------------------------------------------------------------------

typedef __attribute__((ext_vector_type(16))) __bf16 v16bf;
typedef __attribute__((ext_vector_type(8)))  __bf16 v8bf;
typedef __attribute__((ext_vector_type(2)))  __bf16 v2bf;
typedef __attribute__((ext_vector_type(8)))  float  v8f;

#define N_NODES 50000
#define MPAD    50048            // 782 * 64
#define N_EDGES 800000
#define G_DIM   1002
#define KPAD_G  1024
#define D_DIM   256
#define C_DIM   20
#define NPAD_C  32
#define BN_EPS  1e-5f

__device__ __forceinline__ __bf16 f2bf(float f) { return (__bf16)f; }

__device__ __forceinline__ v8f wmma_bf16(v16bf a, v16bf b, v8f c) {
    return __builtin_amdgcn_wmma_f32_16x16x32_bf16(false, a, false, b,
                                                   (short)0, c, false, false);
}

// fragment = 16 bf16 per lane: K = p[0..7] and p[16..23]  (two b128 loads)
__device__ __forceinline__ v16bf frag16(const __bf16* p) {
    v8bf lo = *(const v8bf*)(p);
    v8bf hi = *(const v8bf*)(p + 16);
    return __builtin_shufflevector(lo, hi, 0, 1, 2, 3, 4, 5, 6, 7,
                                           8, 9, 10, 11, 12, 13, 14, 15);
}

// --------------------------------------------------------------------------
// Weight prep: W [K x Ncols] fp32 row-major -> Bt [Npad x Kpad] bf16
// (transposed + zero padded) so B fragments are contiguous b128 loads.
// --------------------------------------------------------------------------
__global__ void prep_b_kernel(const float* __restrict__ W, __bf16* __restrict__ Bt,
                              int K, int Ncols, int Kpad, int Npad) {
    int idx = blockIdx.x * blockDim.x + threadIdx.x;
    if (idx >= Npad * Kpad) return;
    int n = idx / Kpad;
    int k = idx - n * Kpad;
    float v = (n < Ncols && k < K) ? W[(size_t)k * Ncols + n] : 0.0f;
    Bt[idx] = f2bf(v);
}

// x fp32 [N_NODES, 1002] -> bf16 [MPAD, 1024], zero padded rows+cols
__global__ void prep_a_kernel(const float* __restrict__ x, __bf16* __restrict__ xb) {
    long idx = (long)blockIdx.x * blockDim.x + threadIdx.x;
    if (idx >= (long)MPAD * KPAD_G) return;
    int r = (int)(idx >> 10);
    int k = (int)(idx & (KPAD_G - 1));
    float v = (r < N_NODES && k < G_DIM) ? x[(size_t)r * G_DIM + k] : 0.0f;
    xb[idx] = f2bf(v);
}

// --------------------------------------------------------------------------
// Unified WMMA GEMM: A bf16 [MPAD, Kpad] (pad rows zero), Bt bf16 [Npad,Kpad].
// Wave tile 64x32: 8 fp32 accumulators, 8 wmma per 32-wide K step.
// Output: either fp32 [N_NODES, ldc] (+bias) or zero-padded bf16 [MPAD, 256].
// --------------------------------------------------------------------------
__global__ void gemm_wmma_kernel(const __bf16* __restrict__ A,
                                 const __bf16* __restrict__ Bt,
                                 int Kpad, int nGroups, int Nreal, int ldc,
                                 const float* __restrict__ bias,
                                 float* __restrict__ Cf,
                                 __bf16* __restrict__ Cbf) {
    const int mgroups = MPAD / 64;                               // 782
    int wid = blockIdx.x * (blockDim.x >> 5) + (threadIdx.x >> 5);
    wid = __builtin_amdgcn_readfirstlane(wid);                   // SGPR tile id
    if (wid >= mgroups * nGroups) return;
    int lane = threadIdx.x & 31;
    int mg = wid / nGroups, ng = wid - mg * nGroups;
    int mbase = mg * 64, nbase = ng * 32;                        // scalar
    int half = lane >> 4, mr = lane & 15;

    v8f acc[4][2] = {};

    const __bf16* bp0 = Bt + (size_t)(nbase + mr)      * Kpad + half * 8;
    const __bf16* bp1 = Bt + (size_t)(nbase + 16 + mr) * Kpad + half * 8;
    const __bf16* ap  = A  + (size_t)(mbase + mr) * Kpad + half * 8;

    for (int kb = 0; kb < Kpad; kb += 32) {
        // pull next K block toward the caches while this block computes
        if (kb + 32 < Kpad) {
            __builtin_prefetch(ap + kb + 32, 0, 0);
            __builtin_prefetch(bp0 + kb + 32, 0, 0);
        }
        v16bf b0 = frag16(bp0 + kb);
        v16bf b1 = frag16(bp1 + kb);
        #pragma unroll
        for (int t = 0; t < 4; ++t) {
            v16bf af = frag16(ap + (size_t)t * 16 * Kpad + kb);
            acc[t][0] = wmma_bf16(af, b0, acc[t][0]);
            acc[t][1] = wmma_bf16(af, b1, acc[t][1]);
        }
    }

    if (Cbf != nullptr) {
        // bf16 store with zero-filled pad rows: next GEMM reads unguarded
        #pragma unroll
        for (int t = 0; t < 4; ++t) {
            #pragma unroll
            for (int g = 0; g < 2; ++g) {
                int col = nbase + g * 16 + mr;
                #pragma unroll
                for (int r = 0; r < 8; ++r) {
                    int row = mbase + t * 16 + r + half * 8;
                    float v = (row < N_NODES) ? acc[t][g][r] : 0.0f;
                    Cbf[(size_t)row * D_DIM + col] = f2bf(v);
                }
            }
        }
    } else {
        #pragma unroll
        for (int t = 0; t < 4; ++t) {
            if (mbase + t * 16 >= N_NODES) continue;             // scalar guard
            #pragma unroll
            for (int g = 0; g < 2; ++g) {
                int col = nbase + g * 16 + mr;
                if (col >= Nreal) continue;
                float bv = (bias != nullptr) ? bias[col] : 0.0f;
                #pragma unroll
                for (int r = 0; r < 8; ++r) {
                    int row = mbase + t * 16 + r + half * 8;
                    Cf[(size_t)row * ldc + col] = acc[t][g][r] + bv;
                }
            }
        }
    }
}

// --------------------------------------------------------------------------
__global__ void zero_f_kernel(float* __restrict__ p, long n) {
    long i = (long)blockIdx.x * blockDim.x + threadIdx.x;
    if (i < n) p[i] = 0.0f;
}

__global__ void deg_count_kernel(const int* __restrict__ dst, float* __restrict__ deg, int E) {
    int e = blockIdx.x * blockDim.x + threadIdx.x;
    if (e < E) atomicAdd(&deg[dst[e]], 1.0f);
}

__global__ void dinv_kernel(const float* __restrict__ deg, float* __restrict__ dinv,
                            float* __restrict__ invdeg, int n) {
    int i = blockIdx.x * blockDim.x + threadIdx.x;
    if (i < n) {
        float d = deg[i] + 1.0f;
        dinv[i]   = rsqrtf(d);
        invdeg[i] = 1.0f / d;
    }
}

// one block per edge, 256 channels, hardware fp32 atomics into agg
__global__ void aggregate_kernel(const int* __restrict__ src, const int* __restrict__ dst,
                                 const float* __restrict__ dinv,
                                 const float* __restrict__ xw,
                                 float* __restrict__ agg, int E) {
    int e = blockIdx.x;
    if (e >= E) return;
    int c = threadIdx.x;
    int s = src[e], d = dst[e];
    float nrm = dinv[s] * dinv[d];
    atomicAdd(&agg[(size_t)d * D_DIM + c], nrm * xw[(size_t)s * D_DIM + c]);
}

// relu(agg + xw*invdeg + bias) -> padded bf16, 2 channels/thread (packed cvt)
__global__ void selfloop_bias_relu_bf16_kernel(const float* __restrict__ agg,
                                               const float* __restrict__ xw,
                                               const float* __restrict__ invdeg,
                                               const float* __restrict__ bias,
                                               __bf16* __restrict__ out) {
    long p = (long)blockIdx.x * blockDim.x + threadIdx.x;      // pair index
    if (p >= (long)MPAD * (D_DIM / 2)) return;
    long idx = p * 2;
    int r = (int)(idx >> 8);
    int c = (int)(idx & 255);
    float v0 = 0.0f, v1 = 0.0f;
    if (r < N_NODES) {
        float2 a = *(const float2*)(agg + idx);
        float2 w = *(const float2*)(xw + idx);
        float id = invdeg[r];
        v0 = fmaxf(a.x + w.x * id + bias[c],     0.0f);
        v1 = fmaxf(a.y + w.y * id + bias[c + 1], 0.0f);
    }
    v2bf o; o[0] = f2bf(v0); o[1] = f2bf(v1);
    *(v2bf*)(out + idx) = o;
}

// per-column mean / rsqrt(var+eps): one block per column, LDS tree reduce
__global__ void bn_stats_kernel(const float* __restrict__ h, float* __restrict__ mu,
                                float* __restrict__ rstd, int n) {
    __shared__ float ssum[256];
    __shared__ float ssq[256];
    int c = blockIdx.x;
    int tid = threadIdx.x;
    float s = 0.0f, q = 0.0f;
    for (int r = tid; r < n; r += blockDim.x) {
        float v = h[(size_t)r * D_DIM + c];
        s += v; q += v * v;
    }
    ssum[tid] = s; ssq[tid] = q;
    __syncthreads();
    for (int off = 128; off > 0; off >>= 1) {
        if (tid < off) { ssum[tid] += ssum[tid + off]; ssq[tid] += ssq[tid + off]; }
        __syncthreads();
    }
    if (tid == 0) {
        float mean = ssum[0] / (float)n;
        float var  = ssq[0] / (float)n - mean * mean;
        mu[c]   = mean;
        rstd[c] = rsqrtf(var + BN_EPS);
    }
}

// relu(BN(h)) -> padded bf16, 2 channels/thread (packed cvt)
__global__ void bn_apply_relu_bf16_kernel(const float* __restrict__ h,
                                          const float* __restrict__ mu,
                                          const float* __restrict__ rstd,
                                          const float* __restrict__ gamma,
                                          const float* __restrict__ beta,
                                          __bf16* __restrict__ out) {
    long p = (long)blockIdx.x * blockDim.x + threadIdx.x;
    if (p >= (long)MPAD * (D_DIM / 2)) return;
    long idx = p * 2;
    int r = (int)(idx >> 8);
    int c = (int)(idx & 255);
    float v0 = 0.0f, v1 = 0.0f;
    if (r < N_NODES) {
        float2 hv = *(const float2*)(h + idx);
        v0 = fmaxf((hv.x - mu[c])     * rstd[c]     * gamma[c]     + beta[c],     0.0f);
        v1 = fmaxf((hv.y - mu[c + 1]) * rstd[c + 1] * gamma[c + 1] + beta[c + 1], 0.0f);
    }
    v2bf o; o[0] = f2bf(v0); o[1] = f2bf(v1);
    *(v2bf*)(out + idx) = o;
}

__global__ void softmax_kernel(const float* __restrict__ logits, float* __restrict__ out, int n) {
    int r = blockIdx.x * blockDim.x + threadIdx.x;
    if (r >= n) return;
    const float* L = logits + (size_t)r * C_DIM;
    float m = -1e30f;
    #pragma unroll
    for (int c = 0; c < C_DIM; ++c) m = fmaxf(m, L[c]);
    float e[C_DIM];
    float s = 0.0f;
    #pragma unroll
    for (int c = 0; c < C_DIM; ++c) { e[c] = __expf(L[c] - m); s += e[c]; }
    float inv = 1.0f / s;
    #pragma unroll
    for (int c = 0; c < C_DIM; ++c) out[(size_t)r * C_DIM + c] = e[c] * inv;
}

// ---------------------------------------------------------------------------
extern "C" void kernel_launch(void* const* d_in, const int* in_sizes, int n_in,
                              void* d_out, int out_size, void* d_ws, size_t ws_size,
                              hipStream_t stream) {
    const float* x       = (const float*)d_in[0];
    const int*   ei      = (const int*)d_in[1];      // [2, E]
    const float* lin_w   = (const float*)d_in[2];
    const float* conv1_w = (const float*)d_in[3];
    const float* conv1_b = (const float*)d_in[4];
    const float* conv2_w = (const float*)d_in[5];
    const float* conv2_b = (const float*)d_in[6];
    const float* mlp1_w  = (const float*)d_in[7];
    const float* mlp1_b  = (const float*)d_in[8];
    const float* bn_g    = (const float*)d_in[9];
    const float* bn_b    = (const float*)d_in[10];
    const float* mlp2_w  = (const float*)d_in[11];
    const float* mlp2_b  = (const float*)d_in[12];

    const int* e_src = ei;
    const int* e_dst = ei + N_EDGES;
    const int N = N_NODES, E = N_EDGES;

    // ---- workspace carve-out ----
    char* ws = (char*)d_ws;
    size_t off = 0;
    auto alloc = [&](size_t bytes) -> void* {
        void* p = ws + off;
        off = (off + bytes + 255) & ~(size_t)255;
        return p;
    };
    __bf16* bt_lin = (__bf16*)alloc((size_t)D_DIM * KPAD_G * 2);   // [256,1024]
    __bf16* bt_c1  = (__bf16*)alloc((size_t)D_DIM * D_DIM * 2);    // [256,256]
    __bf16* bt_c2  = (__bf16*)alloc((size_t)D_DIM * D_DIM * 2);
    __bf16* bt_m1  = (__bf16*)alloc((size_t)D_DIM * D_DIM * 2);
    __bf16* bt_m2  = (__bf16*)alloc((size_t)NPAD_C * D_DIM * 2);   // [32,256]
    __bf16* xbf    = (__bf16*)alloc((size_t)MPAD * KPAD_G * 2);    // x as bf16 (L2-resident)
    __bf16* abf    = (__bf16*)alloc((size_t)MPAD * D_DIM * 2);     // bf16 activation staging
    float* deg    = (float*)alloc((size_t)N * 4);
    float* dinv   = (float*)alloc((size_t)N * 4);
    float* invdeg = (float*)alloc((size_t)N * 4);
    float* mu     = (float*)alloc((size_t)D_DIM * 4);
    float* rstd   = (float*)alloc((size_t)D_DIM * 4);
    float* fbuf0  = (float*)alloc((size_t)N * D_DIM * 4);
    float* fbuf1  = (float*)alloc((size_t)N * D_DIM * 4);

    const long ND   = (long)N * D_DIM;
    const long NDP2 = (long)MPAD * (D_DIM / 2);     // pair-indexed epilogues
    const int  mgroups = MPAD / 64;                 // 782

    auto gemm = [&](const __bf16* A, const __bf16* Bt, int Kpad, int nGroups,
                    int Nreal, int ldc, const float* bias,
                    float* Cf, __bf16* Cbf) {
        long waves = (long)mgroups * nGroups;
        int blocks = (int)((waves * 32 + 255) / 256);
        gemm_wmma_kernel<<<blocks, 256, 0, stream>>>(A, Bt, Kpad, nGroups,
                                                     Nreal, ldc, bias, Cf, Cbf);
    };

    // ---- 1. weight / input prep ----
    prep_b_kernel<<<(D_DIM * KPAD_G + 255) / 256, 256, 0, stream>>>(lin_w, bt_lin, G_DIM, D_DIM, KPAD_G, D_DIM);
    prep_b_kernel<<<(D_DIM * D_DIM + 255) / 256, 256, 0, stream>>>(conv1_w, bt_c1, D_DIM, D_DIM, D_DIM, D_DIM);
    prep_b_kernel<<<(D_DIM * D_DIM + 255) / 256, 256, 0, stream>>>(conv2_w, bt_c2, D_DIM, D_DIM, D_DIM, D_DIM);
    prep_b_kernel<<<(D_DIM * D_DIM + 255) / 256, 256, 0, stream>>>(mlp1_w, bt_m1, D_DIM, D_DIM, D_DIM, D_DIM);
    prep_b_kernel<<<(NPAD_C * D_DIM + 255) / 256, 256, 0, stream>>>(mlp2_w, bt_m2, D_DIM, C_DIM, D_DIM, NPAD_C);
    {
        long tot = (long)MPAD * KPAD_G;
        prep_a_kernel<<<(int)((tot + 255) / 256), 256, 0, stream>>>(x, xbf);
    }

    // ---- 2. degrees / symmetric GCN normalization ----
    zero_f_kernel<<<(N + 255) / 256, 256, 0, stream>>>(deg, N);
    deg_count_kernel<<<(E + 255) / 256, 256, 0, stream>>>(e_dst, deg, E);
    dinv_kernel<<<(N + 255) / 256, 256, 0, stream>>>(deg, dinv, invdeg, N);

    // ---- 3. encoder: h0 = x @ lin_w  -> abf (bf16) ----
    gemm(xbf, bt_lin, KPAD_G, D_DIM / 32, D_DIM, D_DIM, nullptr, nullptr, abf);

    // ---- 4. conv1 ----
    gemm(abf, bt_c1, D_DIM, D_DIM / 32, D_DIM, D_DIM, nullptr, fbuf0, nullptr);   // xw1
    zero_f_kernel<<<(int)((ND + 255) / 256), 256, 0, stream>>>(fbuf1, ND);
    aggregate_kernel<<<E, D_DIM, 0, stream>>>(e_src, e_dst, dinv, fbuf0, fbuf1, E);
    selfloop_bias_relu_bf16_kernel<<<(int)((NDP2 + 255) / 256), 256, 0, stream>>>(fbuf1, fbuf0, invdeg, conv1_b, abf);

    // ---- 5. conv2 ----
    gemm(abf, bt_c2, D_DIM, D_DIM / 32, D_DIM, D_DIM, nullptr, fbuf0, nullptr);   // xw2
    zero_f_kernel<<<(int)((ND + 255) / 256), 256, 0, stream>>>(fbuf1, ND);
    aggregate_kernel<<<E, D_DIM, 0, stream>>>(e_src, e_dst, dinv, fbuf0, fbuf1, E);
    selfloop_bias_relu_bf16_kernel<<<(int)((NDP2 + 255) / 256), 256, 0, stream>>>(fbuf1, fbuf0, invdeg, conv2_b, abf);

    // ---- 6. mlp1 + bias -> fp32 ----
    gemm(abf, bt_m1, D_DIM, D_DIM / 32, D_DIM, D_DIM, mlp1_b, fbuf0, nullptr);

    // ---- 7. batchnorm (training stats) + relu -> bf16 ----
    bn_stats_kernel<<<D_DIM, 256, 0, stream>>>(fbuf0, mu, rstd, N);
    bn_apply_relu_bf16_kernel<<<(int)((NDP2 + 255) / 256), 256, 0, stream>>>(fbuf0, mu, rstd, bn_g, bn_b, abf);

    // ---- 8. mlp2 + bias -> logits fp32 ----
    gemm(abf, bt_m2, D_DIM, NPAD_C / 32, C_DIM, C_DIM, mlp2_b, fbuf1, nullptr);

    // ---- 9. softmax ----
    softmax_kernel<<<(N + 255) / 256, 256, 0, stream>>>(fbuf1, (float*)d_out, N);
}
